// SwinTransformerBlock_36455682408884
// MI455X (gfx1250) — matrix operand order repaired and checked by reading, not compiled, not run on hardware
//
#include <hip/hip_runtime.h>
#include <hip/hip_bf16.h>
#include <math.h>

typedef unsigned short u16;
typedef __bf16 v16bf __attribute__((ext_vector_type(16)));
typedef __bf16 v8bf  __attribute__((ext_vector_type(8)));
typedef float  v8f   __attribute__((ext_vector_type(8)));
typedef unsigned int uiv8 __attribute__((ext_vector_type(8)));

#define WMMA_BF16(a,b,c) __builtin_amdgcn_wmma_f32_16x16x32_bf16(false,(a),false,(b),(short)0,(c),false,false)

__device__ __forceinline__ u16 f2bf(float f){
    unsigned u = __float_as_uint(f);
    return (u16)((u + 0x7FFFu + ((u >> 16) & 1u)) >> 16);
}
__device__ __forceinline__ float bf2f(u16 h){ return __uint_as_float(((unsigned)h) << 16); }

// Async copy 16B: global -> LDS, tracked by ASYNCcnt (CDNA5 ISA 15.18.3 op 98).
// VDST = LDS byte address (VGPR), VADDR = 64-bit global address (VGPR pair).
__device__ __forceinline__ void async_g2l_b128(void* lds_ptr, const void* gptr)
{
    unsigned lds = (unsigned)(unsigned long long)lds_ptr;   // low 32 bits of flat = LDS offset
    unsigned long long g = (unsigned long long)gptr;
    asm volatile("global_load_async_to_lds_b128 %0, %1, off"
                 :: "v"(lds), "v"(g) : "memory");
}
__device__ __forceinline__ void wait_async0()
{
    asm volatile("s_wait_asynccnt 0x0" ::: "memory");
}

union AFrag { v16bf v; v8bf h[2]; };

// ---------------- weight fp32 (K,N) -> bf16 transposed (N,K) ----------------
__global__ void tconv_w(const float* __restrict__ src, u16* __restrict__ dst, int K, int N)
{
    size_t i = (size_t)blockIdx.x * 256 + threadIdx.x;
    if (i >= (size_t)K * N) return;
    size_t n = i / (size_t)K, k = i % (size_t)K;
    dst[i] = f2bf(src[k * (size_t)N + n]);
}

// ---------------- LN1 + roll(-4,-4) + window partition -> bf16 ----------------
__global__ __launch_bounds__(256)
void ln1_roll_win(const float* __restrict__ x, const float* __restrict__ g,
                  const float* __restrict__ b, u16* __restrict__ out)
{
    const int lane = threadIdx.x & 31, wv = threadIdx.x >> 5;
    const int o = blockIdx.x * 8 + wv;              // windowed token id, 65536 total
    const int bi = o >> 12, rem = o & 4095, win = rem >> 6, np = rem & 63;
    const int hh = (((win >> 3) << 3) + (np >> 3) + 4) & 63;
    const int ww = (((win & 7) << 3) + (np & 7) + 4) & 63;
    const float* src = x + (((size_t)bi << 12) + hh * 64 + ww) * 512;
    float vals[16]; float s = 0.f, s2 = 0.f;
    #pragma unroll
    for (int i = 0; i < 16; ++i) { float v = src[lane + i * 32]; vals[i] = v; s += v; s2 += v * v; }
    #pragma unroll
    for (int off = 16; off >= 1; off >>= 1) { s += __shfl_xor(s, off, 32); s2 += __shfl_xor(s2, off, 32); }
    float mean = s * (1.f / 512.f);
    float var  = s2 * (1.f / 512.f) - mean * mean;
    float rstd = rsqrtf(var + 1e-5f);
    u16* dst = out + (size_t)o * 512;
    #pragma unroll
    for (int i = 0; i < 16; ++i) { int c = lane + i * 32; dst[c] = f2bf((vals[i] - mean) * rstd * g[c] + b[c]); }
}

// ---------------- plain LN (fp32 in, bf16 out) ----------------
__global__ __launch_bounds__(256)
void ln2_plain(const float* __restrict__ x, const float* __restrict__ g,
               const float* __restrict__ b, u16* __restrict__ out)
{
    const int lane = threadIdx.x & 31, wv = threadIdx.x >> 5;
    const int o = blockIdx.x * 8 + wv;
    const float* src = x + (size_t)o * 512;
    float vals[16]; float s = 0.f, s2 = 0.f;
    #pragma unroll
    for (int i = 0; i < 16; ++i) { float v = src[lane + i * 32]; vals[i] = v; s += v; s2 += v * v; }
    #pragma unroll
    for (int off = 16; off >= 1; off >>= 1) { s += __shfl_xor(s, off, 32); s2 += __shfl_xor(s2, off, 32); }
    float mean = s * (1.f / 512.f);
    float var  = s2 * (1.f / 512.f) - mean * mean;
    float rstd = rsqrtf(var + 1e-5f);
    u16* dst = out + (size_t)o * 512;
    #pragma unroll
    for (int i = 0; i < 16; ++i) { int c = lane + i * 32; dst[c] = f2bf((vals[i] - mean) * rstd * g[c] + b[c]); }
}

// ---------------- generic bf16 WMMA GEMM: D = A(MxK) * Bt(NxK)^T + bias ----------------
// Tile staging uses CDNA5 async global->LDS (ASYNCcnt) instead of VGPR round-trips.
// MODE 0: bf16 out + bias
// MODE 1: bf16 out + bias + exact GELU
// MODE 2: fp32 out + bias + residual (same index)
// MODE 3: fp32 out + bias + window-reverse/roll scatter residual (proj, N must be 512)
template<int MODE>
__global__ __launch_bounds__(256)
void gemm_bf16_wmma(const u16* __restrict__ A, const u16* __restrict__ Bt,
                    const float* __restrict__ bias,
                    u16* __restrict__ Dbf, float* __restrict__ Df,
                    const float* __restrict__ Res,
                    int M, int K, int N)
{
    __shared__ __align__(32) u16 As[128 * 32];
    __shared__ __align__(32) u16 Bs[128 * 32];
    const int tid = threadIdx.x, lane = tid & 31;
    const int wv = tid >> 5, wm = wv >> 2, wn = wv & 3;
    const int bm = blockIdx.y * 128, bn = blockIdx.x * 128;
    const int r = tid >> 1, hf = tid & 1;
    const size_t gaRow = (size_t)(bm + r) * K + hf * 16;
    const size_t gbRow = (size_t)(bn + r) * K + hf * 16;
    u16* ldsA = &As[r * 32 + hf * 16];
    u16* ldsB = &Bs[r * 32 + hf * 16];

    v8f acc[4][2] = {};
    for (int k0 = 0; k0 < K; k0 += 32) {
        // async stage: each lane moves 2x16B of A and 2x16B of B straight to LDS
        async_g2l_b128(ldsA,     A  + gaRow + k0);
        async_g2l_b128(ldsA + 8, A  + gaRow + k0 + 8);
        async_g2l_b128(ldsB,     Bt + gbRow + k0);
        async_g2l_b128(ldsB + 8, Bt + gbRow + k0 + 8);
        if (k0 + 32 < K) {
            __builtin_prefetch(A  + gaRow + k0 + 32, 0, 1);
            __builtin_prefetch(Bt + gbRow + k0 + 32, 0, 1);
        }
        wait_async0();
        __syncthreads();
        AFrag af[4]; v16bf bfr[2];
        #pragma unroll
        for (int mt = 0; mt < 4; ++mt) {
            const u16* p = &As[(wm * 64 + mt * 16 + (lane & 15)) * 32 + ((lane >> 4) * 16)];
            af[mt].h[0] = *(const v8bf*)p;
            af[mt].h[1] = *(const v8bf*)(p + 8);
        }
        #pragma unroll
        for (int nt = 0; nt < 2; ++nt)
            bfr[nt] = *(const v16bf*)&Bs[(wn * 32 + nt * 16 + (lane & 15)) * 32 + ((lane >> 4) * 16)];
        #pragma unroll
        for (int mt = 0; mt < 4; ++mt)
            #pragma unroll
            for (int nt = 0; nt < 2; ++nt)
                acc[mt][nt] = WMMA_BF16(af[mt].v, bfr[nt], acc[mt][nt]);
        __syncthreads();
    }

    #pragma unroll
    for (int mt = 0; mt < 4; ++mt) {
        #pragma unroll
        for (int nt = 0; nt < 2; ++nt) {
            #pragma unroll
            for (int i = 0; i < 8; ++i) {
                int m = bm + wm * 64 + mt * 16 + i + ((lane >> 4) << 3);
                int n = bn + wn * 32 + nt * 16 + (lane & 15);
                float v = acc[mt][nt][i] + bias[n];
                if (MODE == 0) {
                    Dbf[(size_t)m * N + n] = f2bf(v);
                } else if (MODE == 1) {
                    float gl = 0.5f * v * (1.0f + erff(v * 0.70710678118654752f));
                    Dbf[(size_t)m * N + n] = f2bf(gl);
                } else if (MODE == 2) {
                    size_t idx = (size_t)m * N + n;
                    Df[idx] = Res[idx] + v;
                } else {
                    int bb = m >> 12, win = (m >> 6) & 63, np = m & 63;
                    int hh = (((win >> 3) << 3) + (np >> 3) + 4) & 63;
                    int ww = (((win & 7) << 3) + (np & 7) + 4) & 63;
                    size_t idx = (((size_t)bb << 12) + hh * 64 + ww) * 512 + n;
                    Df[idx] = Res[idx] + v;
                }
            }
        }
    }
}

// ---------------- windowed top-k attention: one block per (window, head) ----------------
__global__ __launch_bounds__(128)
void attn_topk_wmma(const u16* __restrict__ qkv, const float* __restrict__ rel_bias,
                    u16* __restrict__ out)
{
    __shared__ __align__(32) u16 Qs[64 * 32];
    __shared__ __align__(32) u16 Ks[64 * 32];
    __shared__ __align__(32) u16 Vt[32 * 64];
    __shared__ float S[64 * 64];
    __shared__ __align__(32) u16 Ps[64 * 64];
    __shared__ float relb[225];

    const int tid = threadIdx.x, lane = tid & 31, wv = tid >> 5;
    const int wh = blockIdx.x;
    const int win = wh >> 4, hd = wh & 15;
    const size_t rowbase = (size_t)win * 64 * 1536;

    if (tid < 225) relb[tid] = rel_bias[tid * 16 + hd];

    {   // stage Q, K (row-major keys x 32) and V transposed (32 x keys) into LDS
        int r = tid >> 1, hf = tid & 1;
        uiv8 q = *(const uiv8*)(qkv + rowbase + (size_t)r * 1536 + hd * 32 + hf * 16);
        uiv8 k = *(const uiv8*)(qkv + rowbase + (size_t)r * 1536 + 512 + hd * 32 + hf * 16);
        *(uiv8*)&Qs[r * 32 + hf * 16] = q;
        *(uiv8*)&Ks[r * 32 + hf * 16] = k;
        union { uiv8 v; u16 s[16]; } vv;
        vv.v = *(const uiv8*)(qkv + rowbase + (size_t)r * 1536 + 1024 + hd * 32 + hf * 16);
        #pragma unroll
        for (int j = 0; j < 16; ++j) Vt[(hf * 16 + j) * 64 + r] = vv.s[j];
    }
    __syncthreads();

    {   // S = scale * Q K^T + rel_bias   (wave wv owns row-tile mt = wv)
        const int mt = wv;
        AFrag af;
        const u16* p = &Qs[(mt * 16 + (lane & 15)) * 32 + ((lane >> 4) * 16)];
        af.h[0] = *(const v8bf*)p; af.h[1] = *(const v8bf*)(p + 8);
        #pragma unroll
        for (int nt = 0; nt < 4; ++nt) {
            v8f c = {};
            v16bf bfr = *(const v16bf*)&Ks[(nt * 16 + (lane & 15)) * 32 + ((lane >> 4) * 16)];
            c = WMMA_BF16(af.v, bfr, c);
            #pragma unroll
            for (int i = 0; i < 8; ++i) {
                int mrow = mt * 16 + i + ((lane >> 4) << 3);
                int ncol = nt * 16 + (lane & 15);
                int qy = mrow >> 3, qx = mrow & 7, ky = ncol >> 3, kx = ncol & 7;
                float bv = relb[(qy - ky + 7) * 15 + (qx - kx + 7)];
                S[mrow * 64 + ncol] = c[i] * 0.17677669529663688f + bv;
            }
        }
    }
    __syncthreads();

    // top-48 (= drop 16 smallest) + softmax, one row per thread
    if (tid < 64) {
        float* row = &S[tid * 64];
        float prev = -3.0e38f;
        #pragma unroll 1
        for (int it = 0; it < 16; ++it) {
            float cur = 3.0e38f;
            #pragma unroll 4
            for (int j = 0; j < 64; ++j) { float x = row[j]; if (x > prev && x < cur) cur = x; }
            if (cur > 1.0e37f) break;
            prev = cur;
        }
        float mx = -3.0e38f;
        #pragma unroll 4
        for (int j = 0; j < 64; ++j) { float x = row[j]; if (x > prev) mx = fmaxf(mx, x); }
        float sum = 0.f;
        #pragma unroll 4
        for (int j = 0; j < 64; ++j) {
            float x = row[j];
            float e = (x > prev) ? __expf(x - mx) : 0.f;
            row[j] = e; sum += e;
        }
        float inv = 1.f / sum;
        #pragma unroll 4
        for (int j = 0; j < 64; ++j) Ps[tid * 64 + j] = f2bf(row[j] * inv);
    }
    __syncthreads();

    {   // O = P @ V   (Vt is hd x keys => exactly the B^T layout we need)
        const int mt = wv;
        #pragma unroll
        for (int nt = 0; nt < 2; ++nt) {
            v8f c = {};
            #pragma unroll
            for (int ks = 0; ks < 2; ++ks) {
                AFrag af;
                const u16* p = &Ps[(mt * 16 + (lane & 15)) * 64 + ks * 32 + ((lane >> 4) * 16)];
                af.h[0] = *(const v8bf*)p; af.h[1] = *(const v8bf*)(p + 8);
                v16bf bfr = *(const v16bf*)&Vt[(nt * 16 + (lane & 15)) * 64 + ks * 32 + ((lane >> 4) * 16)];
                c = WMMA_BF16(af.v, bfr, c);
            }
            #pragma unroll
            for (int i = 0; i < 8; ++i) {
                int mrow = mt * 16 + i + ((lane >> 4) << 3);
                int ncol = hd * 32 + nt * 16 + (lane & 15);
                out[((size_t)(win * 64 + mrow)) * 512 + ncol] = f2bf(c[i]);
            }
        }
    }
}

extern "C" void kernel_launch(void* const* d_in, const int* in_sizes, int n_in,
                              void* d_out, int out_size, void* d_ws, size_t ws_size,
                              hipStream_t stream)
{
    const float* x      = (const float*)d_in[0];
    const float* n1g    = (const float*)d_in[1];
    const float* n1b    = (const float*)d_in[2];
    const float* qkv_w  = (const float*)d_in[3];
    const float* qkv_b  = (const float*)d_in[4];
    const float* rel_b  = (const float*)d_in[5];
    const float* proj_w = (const float*)d_in[6];
    const float* proj_b = (const float*)d_in[7];
    const float* n2g    = (const float*)d_in[8];
    const float* n2b    = (const float*)d_in[9];
    const float* fc1_w  = (const float*)d_in[10];
    const float* fc1_b  = (const float*)d_in[11];
    const float* fc2_w  = (const float*)d_in[12];
    const float* fc2_b  = (const float*)d_in[13];
    float* outp = (float*)d_out;

    // workspace layout (bytes)
    char* ws = (char*)d_ws;
    u16* qkv_wT  = (u16*)(ws + 0);           // 1536x512 bf16
    u16* proj_wT = (u16*)(ws + 1572864);     // 512x512
    u16* fc1_wT  = (u16*)(ws + 2097152);     // 2048x512
    u16* fc2_wT  = (u16*)(ws + 4194304);     // 512x2048
    char* RB = ws + 6291456;                 // 256MB multi-use region
    u16* xw   = (u16*)(RB);                  // 65536x512  bf16 (dead after qkv gemm)
    u16* qkvb = (u16*)(RB + 67108864);       // 65536x1536 bf16 (dead after attention)
    u16* h1   = (u16*)(RB);                  // 65536x2048 bf16 (written after both dead)
    char* RC = RB + 268435456;               // 64MB region
    u16* attn_out = (u16*)RC;                // 65536x512 bf16 (dead after proj gemm)
    u16* y2       = (u16*)RC;                // 65536x512 bf16 (written after proj)
    float* x1 = (float*)(RC + 67108864);     // 65536x512 fp32 residual

    // 1) weights to bf16, transposed to (N,K)
    tconv_w<<<(1536 * 512 + 255) / 256, 256, 0, stream>>>(qkv_w,  qkv_wT,  512, 1536);
    tconv_w<<<(512 * 512 + 255) / 256,  256, 0, stream>>>(proj_w, proj_wT, 512, 512);
    tconv_w<<<(512 * 2048 + 255) / 256, 256, 0, stream>>>(fc1_w,  fc1_wT,  512, 2048);
    tconv_w<<<(2048 * 512 + 255) / 256, 256, 0, stream>>>(fc2_w,  fc2_wT,  2048, 512);

    // 2) LN1 + roll + window partition
    ln1_roll_win<<<8192, 256, 0, stream>>>(x, n1g, n1b, xw);

    // 3) QKV projection
    gemm_bf16_wmma<0><<<dim3(12, 512), 256, 0, stream>>>(xw, qkv_wT, qkv_b, qkvb, nullptr, nullptr, 65536, 512, 1536);

    // 4) windowed top-k attention
    attn_topk_wmma<<<16384, 128, 0, stream>>>(qkvb, rel_b, attn_out);

    // 5) output projection + window-reverse/roll scatter + residual -> x1
    gemm_bf16_wmma<3><<<dim3(4, 512), 256, 0, stream>>>(attn_out, proj_wT, proj_b, nullptr, x1, x, 65536, 512, 512);

    // 6) LN2
    ln2_plain<<<8192, 256, 0, stream>>>(x1, n2g, n2b, y2);

    // 7) MLP fc1 + GELU
    gemm_bf16_wmma<1><<<dim3(16, 512), 256, 0, stream>>>(y2, fc1_wT, fc1_b, h1, nullptr, nullptr, 65536, 512, 2048);

    // 8) MLP fc2 + residual -> out
    gemm_bf16_wmma<2><<<dim3(4, 512), 256, 0, stream>>>(h1, fc2_wT, fc2_b, nullptr, outp, x1, 65536, 2048, 512);
}
